// UpSample_2293512536590
// MI455X (gfx1250) — matrix operand order repaired and checked by reading, not compiled, not run on hardware
//
#include <hip/hip_runtime.h>
#include <stdint.h>

// ---------------------------------------------------------------------------
// Polyphase upsample-by-2 FIR (10001 taps) via f16 WMMA, f32 accumulate.
//   out[b][2m+r] = sum_t w[2t+r] * x[b][m-t]
// GEMM tiles: C[i=m,j=batch] = A(filter Toeplitz, d=m0-k0) x B(x slab).
// d is wave-invariant per reduction chunk -> 2 A tiles / 16 WMMAs.
// ---------------------------------------------------------------------------

typedef _Float16 half16 __attribute__((ext_vector_type(16)));
typedef float    float8 __attribute__((ext_vector_type(8)));

// ---- problem constants -----------------------------------------------------
#define BATCH   32
#define LIN     262144
#define FLEN    10001
#define NOUT    534288          // 2*LIN + FLEN - 1
#define NP      267144          // outputs per phase
#define NTAPS0  5001
#define NTAPS1  5000

// ---- tiling constants ------------------------------------------------------
#define PADL    5024            // left zero pad (>= NTAPS0-1, mult of 32)
#define LP      272640          // PADL + LIN + right pad, mult of 16
#define NG      (LP / 16)       // 17040 granules of 16 positions
#define NC      158             // reduction chunks: d = 5024 - 32c  in [0,5024]
#define NADI    158             // precomputed A-tile diagonals (d = 32*adi)
#define MBLK    1024            // m-extent per workgroup (8 waves x 8 tiles x 16)
#define NMB     261             // ceil(NP / MBLK)
#define RING_SLOTS 128          // LDS ring granules (64 KB), power of two
#define PREF_PAIRS 40           // granule-pair prefetch prologue depth
#define ASYNC_SLACK 8           // s_wait_asynccnt bound (rounds of latency slack)

// ---- workspace layout (needs ~17.4 MiB) ------------------------------------
#define ATILES_BYTES ((size_t)NADI * 2 * 1024)   // 158 diag x 2 phases x 1KB
#define XTG_OFF      ((size_t)512 * 1024)
// xtg: NG * 1024 bytes  (granule = 32 batches x 16 halves, batch-major)

// ---- CDNA5 async global->LDS path (probe via __has_builtin) -----------------
#if defined(__has_builtin)
#  if __has_builtin(__builtin_amdgcn_global_load_async_to_lds_b128)
#    define HAVE_ASYNC_LDS 1
#  endif
#endif
#ifndef HAVE_ASYNC_LDS
#  define HAVE_ASYNC_LDS 0
#endif

#if HAVE_ASYNC_LDS
// Param 1 is pointer to <4 x i32> in AS1 (confirmed by round-1 diagnostic).
typedef int v4i __attribute__((vector_size(16)));
typedef __attribute__((address_space(1))) v4i as1_v4i;
typedef __attribute__((address_space(3))) v4i as3_v4i;
#endif

__device__ __forceinline__ void wait_async_slack() {
#if HAVE_ASYNC_LDS
#  if defined(__has_builtin) && __has_builtin(__builtin_amdgcn_s_wait_asynccnt)
  __builtin_amdgcn_s_wait_asynccnt(ASYNC_SLACK);
#  else
  asm volatile("s_wait_asynccnt 8" ::: "memory");
#  endif
#endif
}

// ---------------------------------------------------------------------------
// Prep 1: padded, batch-transposed, B-operand-swizzled f16 copy of x.
// Granule G covers padded positions [16G,16G+16); layout [G][b][k] halves.
// ---------------------------------------------------------------------------
__global__ __launch_bounds__(256) void build_xtg(const float* __restrict__ x,
                                                 uint8_t* __restrict__ xtg) {
  size_t idx = (size_t)blockIdx.x * 256 + threadIdx.x;   // one per f16 element
  if (idx >= (size_t)NG * 512) return;
  int    k   = (int)(idx & 15);
  int    b   = (int)((idx >> 4) & 31);
  size_t G   = idx >> 9;
  long   pos = (long)G * 16 + k - PADL;
  float  v   = (pos >= 0 && pos < LIN) ? x[(size_t)b * LIN + pos] : 0.0f;
  *(_Float16*)(xtg + G * 1024 + b * 32 + k * 2) = (_Float16)v;
}

// ---------------------------------------------------------------------------
// Prep 2: A tiles (16x32 f16 Toeplitz of filter) in this kernel's A-operand lane order (contiguous k, as B).
// Lane l, half h:  M = l&15,  K = h + 16*(l>>4)
// A[M,K] = h_ph[d + M - K], zero outside [0, ntaps_ph).
// ---------------------------------------------------------------------------
__global__ __launch_bounds__(256) void build_atiles(const float* __restrict__ w,
                                                    uint8_t* __restrict__ at) {
  int idx = blockIdx.x * 256 + threadIdx.x;
  if (idx >= NADI * 2 * 512) return;
  int h    = idx & 15;
  int lane = (idx >> 4) & 31;
  int ph   = (idx >> 9) & 1;
  int adi  = idx >> 10;
  int d    = adi * 32;
  int M    = lane & 15;
  int lg   = lane >> 4;
  int K    = h + 16 * lg;
  int t    = d + M - K;
  int ntaps = ph ? NTAPS1 : NTAPS0;
  float v  = (t >= 0 && t < ntaps) ? w[2 * t + ph] : 0.0f;
  *(_Float16*)(at + (size_t)(adi * 2 + ph) * 1024 + lane * 32 + h * 2) = (_Float16)v;
}

// ---------------------------------------------------------------------------
// Main kernel: grid (NMB, 2 batch groups), 256 threads = 8 waves.
// Wave w owns m0 = Mblk + 16w + 128j for j=0..7, both phases (16 accumulators).
// Chunk c: d = 5024-32c (uniform); B granules rel = w + 8j + 2c (+1 hi lanes)
// streamed through a 128-slot LDS ring, async-prefetched PREF_PAIRS ahead.
// ---------------------------------------------------------------------------
__global__ __launch_bounds__(256) void fir_up2_wmma(
    const uint8_t* __restrict__ atiles,
    const uint8_t* __restrict__ xtg,
    float* __restrict__ out) {
  __shared__ __align__(16) uint8_t ring[RING_SLOTS * 512];   // 64 KB

  const int tid  = threadIdx.x;
  const int wave = tid >> 5;
  const int lane = tid & 31;
  const int lg   = lane >> 4;      // 0: K 0..15 / M 0..7   1: K 16..31 / M 8..15
  const int n    = lane & 15;      // batch column within group

  const int    Mblk  = blockIdx.x * MBLK;
  const int    g     = blockIdx.y;             // batch group (16 batches)
  const size_t Gbase = (size_t)(Mblk / 16);    // ring-relative granule origin

  float8 acc0[8] = {};   // phase 0 accumulators (even outputs)
  float8 acc1[8] = {};   // phase 1 accumulators (odd outputs)

  // stage one granule pair p (rel granules 2p,2p+1): threads 0..63, 16B each
  auto stage_pair = [&](int p) {
    if (tid < 64) {
      const int Gr  = 2 * p + (tid >> 5);
      const int seg = tid & 31;
      const uint8_t* src = xtg + (Gbase + (size_t)Gr) * 1024u
                               + (size_t)g * 512u + (size_t)seg * 16u;
      uint8_t* dst = &ring[((Gr & (RING_SLOTS - 1)) * 512) + seg * 16];
#if HAVE_ASYNC_LDS
      __builtin_amdgcn_global_load_async_to_lds_b128(
          (as1_v4i*)(uintptr_t)src, (as3_v4i*)(uint32_t)(uintptr_t)dst, 0, 0);
#else
      *(uint4*)dst = *(const uint4*)src;
#endif
    }
  };

  // prologue: fill initial window (80 granules)
  for (int p = 0; p < PREF_PAIRS; ++p) stage_pair(p);

  // LDS base for this wave's B tiles: granule rel = wave + 8j + 2c + lg
  const int bofs = n * 32;

  for (int c = 0; c < NC; ++c) {
    stage_pair(c + PREF_PAIRS);   // prefetch; rel <= 2(NC-1+PREF)+1 = 395 < NG-Gbase
    wait_async_slack();           // pairs <= c+32 complete (covers rel 2c+65)
    __syncthreads();              // publish staged granules to all waves

    const int adi = (NC - 1) - c;                       // d = 32*adi
    const uint8_t* a0p = atiles + (size_t)(adi * 2 + 0) * 1024u + lane * 32;
    const uint8_t* a1p = atiles + (size_t)(adi * 2 + 1) * 1024u + lane * 32;
    const half16 A0 = *(const half16*)a0p;              // 2x global_load_b128
    const half16 A1 = *(const half16*)a1p;
    if (c + 1 < NC) {
      __builtin_prefetch(a0p - 2048, 0, 1);             // next diagonal's tiles
      __builtin_prefetch(a1p - 2048, 0, 1);
    }

    // B double-buffer: keep next tile's ds_loads in flight across current WMMAs
    const int rel0 = wave + 2 * c + lg;
    half16 Bcur = *(const half16*)&ring[((rel0 & (RING_SLOTS - 1)) * 512) + bofs];
#pragma unroll
    for (int j = 0; j < 8; ++j) {
      half16 Bnxt = Bcur;
      if (j < 7) {
        const int rel = rel0 + 8 * (j + 1);
        Bnxt = *(const half16*)&ring[((rel & (RING_SLOTS - 1)) * 512) + bofs];
      }
      acc0[j] = __builtin_amdgcn_wmma_f32_16x16x32_f16(
          false, A0, false, Bcur, (short)0, acc0[j], false, false);
      acc1[j] = __builtin_amdgcn_wmma_f32_16x16x32_f16(
          false, A1, false, Bcur, (short)0, acc1[j], false, false);
      Bcur = Bnxt;
    }
  }

  // store: interleave the two phases -> contiguous even/odd output pairs
#pragma unroll
  for (int j = 0; j < 8; ++j) {
    const int m0 = Mblk + 16 * wave + 128 * j;
    const int b  = g * 16 + n;
    float* rowp  = out + (size_t)b * NOUT;
#pragma unroll
    for (int v = 0; v < 8; ++v) {
      const int m = m0 + v + 8 * lg;                    // D layout: M = v + 8*lg
      if (m < NP) {
        float2 val;
        val.x = acc0[j][v];                             // out[b][2m]
        val.y = acc1[j][v];                             // out[b][2m+1]
        *(float2*)(rowp + 2 * (size_t)m) = val;
      }
    }
  }
}

// ---------------------------------------------------------------------------
extern "C" void kernel_launch(void* const* d_in, const int* in_sizes, int n_in,
                              void* d_out, int out_size, void* d_ws, size_t ws_size,
                              hipStream_t stream) {
  (void)in_sizes; (void)n_in; (void)out_size; (void)ws_size;
  const float* x = (const float*)d_in[0];   // (32, 262144) f32
  const float* w = (const float*)d_in[1];   // (10001,) f32 (already * factor)
  float* out     = (float*)d_out;           // (32, 534288) f32
  uint8_t* ws    = (uint8_t*)d_ws;          // needs ~17.4 MiB
  uint8_t* at    = ws;                      // A tiles: 316 KB
  uint8_t* xtg   = ws + XTG_OFF;            // x granules: ~16.7 MiB

  {
    size_t total = (size_t)NG * 512;
    int blocks = (int)((total + 255) / 256);             // 34080
    build_xtg<<<blocks, 256, 0, stream>>>(x, xtg);
  }
  {
    int blocks = (NADI * 2 * 512 + 255) / 256;           // 632
    build_atiles<<<blocks, 256, 0, stream>>>(w, at);
  }
  {
    dim3 grid(NMB, 2);                                   // 261 m-blocks x 2 batch groups
    fir_up2_wmma<<<grid, 256, 0, stream>>>(at, xtg, out);
  }
}
